// AdaptiveCSABlock_30528627540333
// MI455X (gfx1250) — matrix operand order repaired and checked by reading, not compiled, not run on hardware
//
#include <hip/hip_runtime.h>
#include <hip/hip_bf16.h>

// ---------- WMMA types & helpers (CDNA5 / gfx1250, wave32) ----------
typedef __attribute__((ext_vector_type(16))) _Float16 v16h;
typedef __attribute__((ext_vector_type(8)))  _Float16 h8;
typedef __attribute__((ext_vector_type(8)))  float    v8f;
typedef __attribute__((ext_vector_type(4)))  unsigned int v4u;
typedef __attribute__((ext_vector_type(8)))  int      v8i;
typedef __attribute__((ext_vector_type(4)))  int      v4i;

__device__ __forceinline__ v8f wmma_f16(v16h a, v16h b, v8f c) {
  return __builtin_amdgcn_wmma_f32_16x16x32_f16(false, a, false, b, (short)0, c, false, false);
}

// A-fragment: 16x32 f16, row-major source with row pointer `row` (f16 units).
// lane m = l&15 selects row (caller bakes it into `row`), hh = l>>4.
// Contiguous layout: hh=0 -> K{0..15}, hh=1 -> K{16..31} (offset by k0).
__device__ __forceinline__ v16h ldA(const _Float16* row, int k0, int hh) {
  h8 lo = *(const h8*)(row + k0 + hh * 16);
  h8 hi = *(const h8*)(row + k0 + hh * 16 + 8);
  return __builtin_shufflevector(lo, hi, 0,1,2,3,4,5,6,7,8,9,10,11,12,13,14,15);
}

// B-fragment: 32x16 f16. Column n (=lane&15) must be K-contiguous at p.
__device__ __forceinline__ v16h ldB(const _Float16* p) {
  h8 lo = *(const h8*)p;
  h8 hi = *(const h8*)(p + 8);
  return __builtin_shufflevector(lo, hi, 0,1,2,3,4,5,6,7,8,9,10,11,12,13,14,15);
}

// ---------- problem constants ----------
#define BB   8
#define CH   192
#define HH_  224
#define WW_  224
#define WSZ  7
#define NHD  6
#define HDIM 32
#define NWIN 8192          // 8 * 32 * 32
#define NTOK 64            // 49 padded to 64
#define NREAL 49

// sXW layout written by the TDM: 192 f16 per row + 8 f16 pad inserted after every
// 64 f16 (pad_interval=32 dwords, pad_amount=4 dwords) -> row stride 216 f16 (432B).
#define SXW_STRIDE 216

// ---------- workspace layout (bytes) ----------
#define OFF_XW    ((size_t)0)                    // f16 [8192][64][192]
#define OFF_XT    ((size_t)201326592)            // f32 [B,H,W,C]
#define OFF_Y1    ((size_t)509607936)            // f32 [B,H,W,C]
#define OFF_QKVB  ((size_t)817889280)            // f16 216 frags * 512
#define OFF_PROJB ((size_t)818110464)            // f16 72 frags
#define OFF_FC1B  ((size_t)818184192)            // f16 288 frags
#define OFF_FC2B  ((size_t)818479104)            // f16 288 frags
#define OFF_BM    ((size_t)818774016)            // f32 [6][64][64]

// ---------- pack fp32 weight [K][N] into WMMA B-fragment layout ----------
__global__ void k_packB(const float* __restrict__ W, _Float16* __restrict__ out,
                        int K, int N) {
  int KT = K >> 5;
  int nfrag = (N >> 4) * KT;
  int t = blockIdx.x * blockDim.x + threadIdx.x;
  if (t >= nfrag * 32) return;
  int f = t >> 5, l = t & 31;
  int nt = f / KT, kt = f - nt * KT;
  int n = nt * 16 + (l & 15);
  int kbase = kt * 32 + (l >> 4) * 16;
  _Float16* d = out + (size_t)f * 512 + l * 16;
  for (int e = 0; e < 16; ++e)
    d[e] = (_Float16)W[(size_t)(kbase + e) * N + n];
}

// ---------- padded relative-position bias matrix [6][64][64] ----------
__global__ void k_bias(const float* __restrict__ tab, const int* __restrict__ ridx,
                       float* __restrict__ bm) {
  int t = blockIdx.x * blockDim.x + threadIdx.x;
  if (t >= NHD * 64 * 64) return;
  int n = t & 63, m = (t >> 6) & 63, hd = t >> 12;
  float v = -1e30f;
  if (m < NREAL && n < NREAL) v = tab[ridx[m * NREAL + n] * NHD + hd];
  bm[t] = v;
}

// ---------- zero the 15 pad token rows of every window ----------
__global__ void k_padzero(unsigned int* __restrict__ xw32) {
  size_t base = (size_t)blockIdx.x * 6144 + NREAL * 96;   // 192 f16 = 96 u32 per row
  for (int i = threadIdx.x; i < 15 * 96; i += blockDim.x) xw32[base + i] = 0u;
}

// ---------- LN1 + window partition (f16) + channel-last x copy ----------
__global__ void __launch_bounds__(256) k_ln1_part(
    const float* __restrict__ x, const float* __restrict__ lw,
    const float* __restrict__ lb, _Float16* __restrict__ xw,
    float* __restrict__ xt) {
  __shared__ float tile[CH * 32];
  __shared__ float smu[32], srs[32];
  int blk = blockIdx.x;
  int w0 = (blk % 7) * 32;
  int h  = (blk / 7) % HH_;
  int b  = blk / (7 * HH_);
  int tid = threadIdx.x;
  const float* xb = x + ((size_t)b * CH) * HH_ * WW_ + (size_t)h * WW_ + w0;
  for (int i = tid; i < CH * 32; i += 256) {
    int c = i >> 5, w = i & 31;
    tile[c * 32 + w] = xb[(size_t)c * HH_ * WW_ + w];
  }
  __syncthreads();
  if (tid < 32) {
    float s = 0.f, s2 = 0.f;
    for (int c = 0; c < CH; ++c) { float v = tile[c * 32 + tid]; s += v; s2 += v * v; }
    float mu = s * (1.f / CH);
    float var = s2 * (1.f / CH) - mu * mu;
    smu[tid] = mu; srs[tid] = rsqrtf(var + 1e-5f);
  }
  __syncthreads();
  int pix = tid >> 3, chk = tid & 7;
  int w = w0 + pix;
  int wh = h / WSZ, ww = w / WSZ;
  int tok = (h % WSZ) * WSZ + (w % WSZ);
  size_t win = ((size_t)b * 32 + wh) * 32 + ww;
  _Float16* dst = xw + win * 12288 + (size_t)tok * CH;
  float* dt = xt + (((size_t)b * HH_ + h) * WW_ + w) * CH;
  float mu = smu[pix], rs = srs[pix];
  for (int j = 0; j < 24; ++j) {
    int c = chk * 24 + j;
    float v = tile[c * 32 + pix];
    dt[c] = v;
    dst[c] = (_Float16)((v - mu) * rs * lw[c] + lb[c]);
  }
}

// ---------- window attention: TDM stage -> QKV GEMM -> QK^T -> softmax -> @V -> proj ----
// one block per window; 6 waves (192 threads); dynamic LDS ~157KB of the 320KB WGP pool
__global__ void __launch_bounds__(192) k_attn(
    const _Float16* __restrict__ xw, const _Float16* __restrict__ qkvB,
    const float* __restrict__ qkv_b, const _Float16* __restrict__ projB,
    const float* __restrict__ proj_b, const float* __restrict__ bm,
    const float* __restrict__ xt, float* __restrict__ y1) {
  extern __shared__ char smem[];
  _Float16* sXW  = (_Float16*)(smem);            // [64][216] A-data (TDM padded); reused as attn_out [64][200]
  _Float16* sQKV = (_Float16*)(smem + 27648);    // [64][392]  q: 0..191, k: 192..383
  _Float16* sVT  = (_Float16*)(smem + 77824);    // [6][32][72] v transposed
  _Float16* sATT = (_Float16*)(smem + 105472);   // [6][64][72] attn probs

  const int tid = threadIdx.x;
  const int wid = __builtin_amdgcn_readfirstlane(tid >> 5);  // wave-uniform scalar
  const int l   = tid & 31;
  const int ln  = l & 15;
  const int hh  = l >> 4;
  const size_t win = blockIdx.x;

  // ---- stage window A-data into LDS ----
#if __has_builtin(__builtin_amdgcn_tensor_load_to_lds) && __has_builtin(__builtin_amdgcn_s_wait_tensorcnt)
  if (wid == 0) {
    // D# group0: count=1 | lds_addr | global_addr | type=2
    unsigned long long ga = (unsigned long long)(const void*)(xw + win * 12288);
    unsigned int lds_off = (unsigned int)(size_t)(void*)sXW;
    v4u g0 = { 1u, lds_off, (unsigned int)ga,
               (unsigned int)(ga >> 32) | (2u << 30) };
    // D# group1: data_size=2B, pad_enable, pad_interval=32dw(code4), pad_amount=4dw(code3)
    // tensor_dim0 = tile_dim0 = 12288 elements, tensor_dim1 = tile_dim1 = 1
    v8i g1 = { (int)((1u << 16) | (1u << 20) | (4u << 22) | (3u << 25)),
               (int)(12288u << 16),      // tensor_dim0[15:0] in bits 63:48
               (int)(0x00010000u),       // tensor_dim0 hi=0, tensor_dim1 lo=1
               (int)(12288u << 16),      // tensor_dim1 hi=0, tile_dim0=12288
               1,                        // tile_dim1=1, tile_dim2=0
               12288,                    // tensor_dim0_stride lo
               0, 0 };
    v4i gz = { 0, 0, 0, 0 };
#if __clang_major__ >= 23
    v8i gz8 = { 0, 0, 0, 0, 0, 0, 0, 0 };
    __builtin_amdgcn_tensor_load_to_lds(g0, g1, gz, gz, gz8, 0);
#else
    __builtin_amdgcn_tensor_load_to_lds(g0, g1, gz, gz, 0);
#endif
    __builtin_amdgcn_s_wait_tensorcnt(0);
  }
#else
  {
    const h8* src = (const h8*)(xw + win * 12288);
    for (int i = tid; i < 1536; i += 192) {
      int row = i / 24, c0 = (i % 24) * 8;
      *(h8*)(sXW + row * SXW_STRIDE + c0 + ((c0 >> 6) * 8)) = src[i];
    }
  }
#endif
  __syncthreads();

  // ---- QKV GEMM: [64,192] x [192,576], 144 tile jobs over 6 waves ----
  for (int job = wid; job < 144; job += 6) {
    int mt = job & 3, nt = job >> 2;
    const _Float16* arow = sXW + (mt * 16 + ln) * SXW_STRIDE;
    v8f acc = {0.f,0.f,0.f,0.f,0.f,0.f,0.f,0.f};
    for (int kt = 0; kt < 6; ++kt) {
      int k0 = kt * 32;
      v16h a = ldA(arow + ((k0 >> 6) * 8), k0, hh);   // skip intra-row TDM pads
      v16h b = *(const v16h*)(qkvB + ((size_t)(nt * 6 + kt)) * 512 + l * 16);
      acc = wmma_f16(a, b, acc);
    }
    int n = nt * 16 + ln;
    float bias = qkv_b[n];
    if (nt < 12) {                       // Q (pre-scaled by 1/sqrt(HD))
      for (int e = 0; e < 8; ++e) {
        int m = mt * 16 + hh * 8 + e;
        sQKV[m * 392 + n] = (_Float16)((acc[e] + bias) * 0.17677669529663687f);
      }
    } else if (nt < 24) {                // K
      for (int e = 0; e < 8; ++e) {
        int m = mt * 16 + hh * 8 + e;
        sQKV[m * 392 + n] = (_Float16)(acc[e] + bias);
      }
    } else {                             // V -> transposed [head][hd][token]
      int hd = (n - 384) >> 5, c = (n - 384) & 31;
      for (int e = 0; e < 8; ++e) {
        int m = mt * 16 + hh * 8 + e;
        sVT[(hd * 32 + c) * 72 + m] = (_Float16)(acc[e] + bias);
      }
    }
  }
  __syncthreads();

  // ---- per-head attention scores + softmax (wave wid == head) ----
  {
    const int hd = wid;
    const float* bmh = bm + hd * 4096;
    for (int mt = 0; mt < 4; ++mt) {
      const _Float16* qrow = sQKV + (mt * 16 + ln) * 392 + hd * 32;
      v8f s[4];
      for (int nt = 0; nt < 4; ++nt) {
        v16h a = ldA(qrow, 0, hh);
        v16h b = ldB(sQKV + (nt * 16 + ln) * 392 + 192 + hd * 32 + hh * 16);
        v8f acc = {0.f,0.f,0.f,0.f,0.f,0.f,0.f,0.f};
        acc = wmma_f16(a, b, acc);
        for (int e = 0; e < 8; ++e)
          acc[e] += bmh[(mt * 16 + hh * 8 + e) * 64 + nt * 16 + ln];
        s[nt] = acc;
      }
      float mx[8];
      for (int e = 0; e < 8; ++e) {
        float m0 = fmaxf(fmaxf(s[0][e], s[1][e]), fmaxf(s[2][e], s[3][e]));
        for (int d = 1; d < 16; d <<= 1) m0 = fmaxf(m0, __shfl_xor(m0, d, 32));
        mx[e] = m0;
      }
      float inv[8];
      for (int e = 0; e < 8; ++e) {
        float t = 0.f;
        for (int nt = 0; nt < 4; ++nt) {
          float p = __expf(s[nt][e] - mx[e]);
          s[nt][e] = p; t += p;
        }
        for (int d = 1; d < 16; d <<= 1) t += __shfl_xor(t, d, 32);
        inv[e] = 1.f / t;
      }
      for (int nt = 0; nt < 4; ++nt)
        for (int e = 0; e < 8; ++e)
          sATT[(hd * 64 + mt * 16 + hh * 8 + e) * 72 + nt * 16 + ln] =
              (_Float16)(s[nt][e] * inv[e]);
    }
  }
  __syncthreads();

  // ---- attn @ V -> attn_out (reuse sXW as [64][200] f16) ----
  {
    const int hd = wid;
    for (int mt = 0; mt < 4; ++mt) {
      const _Float16* arow = sATT + (hd * 64 + mt * 16 + ln) * 72;
      for (int nt = 0; nt < 2; ++nt) {
        v8f acc = {0.f,0.f,0.f,0.f,0.f,0.f,0.f,0.f};
        for (int kt = 0; kt < 2; ++kt) {
          v16h a = ldA(arow, kt * 32, hh);
          v16h b = ldB(sVT + (hd * 32 + nt * 16 + ln) * 72 + kt * 32 + hh * 16);
          acc = wmma_f16(a, b, acc);
        }
        int col = hd * 32 + nt * 16 + ln;
        for (int e = 0; e < 8; ++e)
          sXW[(mt * 16 + hh * 8 + e) * 200 + col] = (_Float16)acc[e];
      }
    }
  }
  __syncthreads();

  // ---- proj GEMM + residual -> y1 [B,H,W,C] ----
  const int b  = (int)(blockIdx.x >> 10);
  const int wh = (int)((blockIdx.x >> 5) & 31);
  const int ww = (int)(blockIdx.x & 31);
  for (int job = wid; job < 48; job += 6) {
    int mt = job & 3, nt = job >> 2;
    const _Float16* arow = sXW + (mt * 16 + ln) * 200;
    v8f acc = {0.f,0.f,0.f,0.f,0.f,0.f,0.f,0.f};
    for (int kt = 0; kt < 6; ++kt) {
      v16h a = ldA(arow, kt * 32, hh);
      v16h b = *(const v16h*)(projB + ((size_t)(nt * 6 + kt)) * 512 + l * 16);
      acc = wmma_f16(a, b, acc);
    }
    int c = nt * 16 + ln;
    float pb = proj_b[c];
    for (int e = 0; e < 8; ++e) {
      int tok = mt * 16 + hh * 8 + e;
      if (tok < NREAL) {
        int h = wh * WSZ + tok / WSZ, w = ww * WSZ + tok % WSZ;
        size_t pix = ((size_t)b * HH_ + h) * WW_ + w;
        y1[pix * CH + c] = xt[pix * CH + c] + acc[e] + pb;
      }
    }
  }
}

// ---------- LN2 + MLP (fc1/gelu/fc2) + residual, output [B,C,H,W] ----------
__global__ void __launch_bounds__(192) k_mlp(
    const float* __restrict__ y1, const float* __restrict__ lw,
    const float* __restrict__ lb, const _Float16* __restrict__ fc1B,
    const float* __restrict__ fc1_b, const _Float16* __restrict__ fc2B,
    const float* __restrict__ fc2_b, float* __restrict__ out) {
  extern __shared__ char smem[];
  _Float16* sXA = (_Float16*)(smem);           // [64][200]
  _Float16* sH1 = (_Float16*)(smem + 25600);   // [64][776]
  const int tid = threadIdx.x;
  const int wid = __builtin_amdgcn_readfirstlane(tid >> 5);
  const int l   = tid & 31;
  const int ln  = l & 15;
  const int hh  = l >> 4;
  const int b  = (int)(blockIdx.x >> 10);
  const int wh = (int)((blockIdx.x >> 5) & 31);
  const int ww = (int)(blockIdx.x & 31);

  // LN2, wave per token
  for (int t = wid; t < NREAL; t += 6) {
    int h = wh * WSZ + t / WSZ, w = ww * WSZ + t % WSZ;
    const float* yrow = y1 + (((size_t)b * HH_ + h) * WW_ + w) * CH;
    float s = 0.f, s2 = 0.f;
    for (int c = l; c < CH; c += 32) { float v = yrow[c]; s += v; s2 += v * v; }
    for (int d = 1; d < 32; d <<= 1) { s += __shfl_xor(s, d, 32); s2 += __shfl_xor(s2, d, 32); }
    float mu = s * (1.f / CH), rs = rsqrtf(s2 * (1.f / CH) - mu * mu + 1e-5f);
    for (int c = l; c < CH; c += 32)
      sXA[t * 200 + c] = (_Float16)((yrow[c] - mu) * rs * lw[c] + lb[c]);
  }
  for (int i = tid; i < 15 * CH; i += 192) {     // zero pad rows 49..63
    int r = NREAL + i / CH, c = i % CH;
    sXA[r * 200 + c] = (_Float16)0.f;
  }
  __syncthreads();

  // fc1 + exact GELU
  for (int job = wid; job < 192; job += 6) {
    int mt = job & 3, nt = job >> 2;             // nt 0..47
    const _Float16* arow = sXA + (mt * 16 + ln) * 200;
    v8f acc = {0.f,0.f,0.f,0.f,0.f,0.f,0.f,0.f};
    for (int kt = 0; kt < 6; ++kt) {
      v16h a = ldA(arow, kt * 32, hh);
      v16h b = *(const v16h*)(fc1B + ((size_t)(nt * 6 + kt)) * 512 + l * 16);
      acc = wmma_f16(a, b, acc);
    }
    int n = nt * 16 + ln;
    float fb = fc1_b[n];
    for (int e = 0; e < 8; ++e) {
      float v = acc[e] + fb;
      v = 0.5f * v * (1.f + erff(v * 0.7071067811865475f));
      sH1[(mt * 16 + hh * 8 + e) * 776 + n] = (_Float16)v;
    }
  }
  __syncthreads();

  // fc2 + residual, transpose back to [B,C,H,W]
  for (int job = wid; job < 48; job += 6) {
    int mt = job & 3, nt = job >> 2;             // nt 0..11
    const _Float16* arow = sH1 + (mt * 16 + ln) * 776;
    v8f acc = {0.f,0.f,0.f,0.f,0.f,0.f,0.f,0.f};
    for (int kt = 0; kt < 24; ++kt) {
      v16h a = ldA(arow, kt * 32, hh);
      v16h b = *(const v16h*)(fc2B + ((size_t)(nt * 24 + kt)) * 512 + l * 16);
      acc = wmma_f16(a, b, acc);
    }
    int c = nt * 16 + ln;
    float fb = fc2_b[c];
    for (int e = 0; e < 8; ++e) {
      int tok = mt * 16 + hh * 8 + e;
      if (tok < NREAL) {
        int h = wh * WSZ + tok / WSZ, w = ww * WSZ + tok % WSZ;
        size_t pix = ((size_t)b * HH_ + h) * WW_ + w;
        out[(((size_t)b * CH + c) * HH_ + h) * WW_ + w] = y1[pix * CH + c] + acc[e] + fb;
      }
    }
  }
}

// ---------- launcher ----------
extern "C" void kernel_launch(void* const* d_in, const int* in_sizes, int n_in,
                              void* d_out, int out_size, void* d_ws, size_t ws_size,
                              hipStream_t stream) {
  const float* x      = (const float*)d_in[0];
  const float* ln1_w  = (const float*)d_in[1];
  const float* ln1_b  = (const float*)d_in[2];
  const float* qkv_w  = (const float*)d_in[3];
  const float* qkv_b  = (const float*)d_in[4];
  const float* proj_w = (const float*)d_in[5];
  const float* proj_b = (const float*)d_in[6];
  const float* rpb    = (const float*)d_in[7];
  const float* ln2_w  = (const float*)d_in[8];
  const float* ln2_b  = (const float*)d_in[9];
  const float* fc1_w  = (const float*)d_in[10];
  const float* fc1_b  = (const float*)d_in[11];
  const float* fc2_w  = (const float*)d_in[12];
  const float* fc2_b  = (const float*)d_in[13];
  const int*   relidx = (const int*)d_in[14];

  char* ws = (char*)d_ws;
  _Float16* xw    = (_Float16*)(ws + OFF_XW);
  float*    xt    = (float*)(ws + OFF_XT);
  float*    y1    = (float*)(ws + OFF_Y1);
  _Float16* qkvB  = (_Float16*)(ws + OFF_QKVB);
  _Float16* projB = (_Float16*)(ws + OFF_PROJB);
  _Float16* fc1B  = (_Float16*)(ws + OFF_FC1B);
  _Float16* fc2B  = (_Float16*)(ws + OFF_FC2B);
  float*    bm    = (float*)(ws + OFF_BM);

  // weight packing (tiny; L2-resident afterwards)
  k_packB<<<(216 * 32 + 255) / 256, 256, 0, stream>>>(qkv_w, qkvB, 192, 576);
  k_packB<<<( 72 * 32 + 255) / 256, 256, 0, stream>>>(proj_w, projB, 192, 192);
  k_packB<<<(288 * 32 + 255) / 256, 256, 0, stream>>>(fc1_w, fc1B, 192, 768);
  k_packB<<<(288 * 32 + 255) / 256, 256, 0, stream>>>(fc2_w, fc2B, 768, 192);
  k_bias<<<(NHD * 64 * 64 + 255) / 256, 256, 0, stream>>>(rpb, relidx, bm);
  k_padzero<<<NWIN, 128, 0, stream>>>((unsigned int*)xw);

  // LN1 + window partition
  k_ln1_part<<<BB * HH_ * 7, 256, 0, stream>>>(x, ln1_w, ln1_b, xw, xt);

  // attention (one block per window; ~157KB dynamic LDS)
  k_attn<<<NWIN, 192, 160768, stream>>>(xw, qkvB, qkv_b, projB, proj_b, bm, xt, y1);

  // MLP (one block per window; ~122KB dynamic LDS)
  k_mlp<<<NWIN, 192, 124928, stream>>>(y1, ln2_w, ln2_b, fc1B, fc1_b, fc2B, fc2_b,
                                       (float*)d_out);
}
